// MLA_88081189306699
// MI455X (gfx1250) — matrix operand order repaired and checked by reading, not compiled, not run on hardware
//
#include <hip/hip_runtime.h>
#include <hip/hip_bf16.h>
#include <cstdint>

// MLA forward for gfx1250 (MI455X), wave32 + v_wmma_f32_16x16x32_f16 +
// global_load_async_to_lds_b128 double-buffered staging.
// B=2, T=2048, D=2048, H=16, HD=128, L=64 hardcoded.

typedef _Float16 half8  __attribute__((ext_vector_type(8)));
typedef _Float16 half16 __attribute__((ext_vector_type(16)));
typedef float    f32x8  __attribute__((ext_vector_type(8)));

#define CAT8_8(lo, hi) __builtin_shufflevector((lo), (hi), 0,1,2,3,4,5,6,7,8,9,10,11,12,13,14,15)

static __device__ __forceinline__ f32x8 wmma32(half16 a, half16 b, f32x8 c) {
  // D = A(16x32 f16) * B(32x16 f16) + C(16x16 f32)
  return __builtin_amdgcn_wmma_f32_16x16x32_f16(false, a, false, b, (short)0, c, false, false);
}
static __device__ __forceinline__ float hmax16(float v) {
#pragma unroll
  for (int m = 1; m < 16; m <<= 1) v = fmaxf(v, __shfl_xor(v, m, 32));
  return v;
}
static __device__ __forceinline__ float hsum16(float v) {
#pragma unroll
  for (int m = 1; m < 16; m <<= 1) v += __shfl_xor(v, m, 32);
  return v;
}

// Async global->LDS copy of one 128x32 f16 slice (8 KB = 512 x 16B chunks,
// 2 chunks per thread => 2 async instructions per wave per slice).
static __device__ __forceinline__ void async_stage(const _Float16* __restrict__ gbase,
                                                   int K, int kk,
                                                   _Float16* lds, int tid) {
#pragma unroll
  for (int it = 0; it < 2; ++it) {
    int c = tid + it * 256;            // chunk id in [0,512)
    int row = c >> 2, q = c & 3;
    const _Float16* g = gbase + (size_t)row * K + kk + q * 8;
    uint32_t loff = (uint32_t)(uintptr_t)(lds + row * 32 + q * 8);  // wave-relative LDS addr
    asm volatile("global_load_async_to_lds_b128 %0, %1, off"
                 :: "v"(loff), "v"((uint64_t)(uintptr_t)g)
                 : "memory");
  }
}

// ---------------------------------------------------------------- LayerNorm
__global__ __launch_bounds__(256) void mla_layernorm(const float* __restrict__ x,
    const float* __restrict__ w, const float* __restrict__ b, _Float16* __restrict__ out) {
  const int row = blockIdx.x;          // B*T rows
  const int tid = threadIdx.x;
  const float* xr = x + (size_t)row * 2048;
  __shared__ float sb[256];
  float v[8];
  float s = 0.f, ss = 0.f;
#pragma unroll
  for (int i = 0; i < 8; i++) { v[i] = xr[tid + 256 * i]; s += v[i]; ss += v[i] * v[i]; }
  sb[tid] = s; __syncthreads();
  for (int o = 128; o > 0; o >>= 1) { if (tid < o) sb[tid] += sb[tid + o]; __syncthreads(); }
  float mean = sb[0] * (1.f / 2048.f); __syncthreads();
  sb[tid] = ss; __syncthreads();
  for (int o = 128; o > 0; o >>= 1) { if (tid < o) sb[tid] += sb[tid + o]; __syncthreads(); }
  float var = sb[0] * (1.f / 2048.f) - mean * mean;
  float rstd = rsqrtf(var + 1e-5f);
  _Float16* orow = out + (size_t)row * 2048;
#pragma unroll
  for (int i = 0; i < 8; i++) {
    int c = tid + 256 * i;
    orow[c] = (_Float16)((v[i] - mean) * rstd * w[c] + b[c]);
  }
}

// ---------------------------------------------- weight convert+transpose f32->f16
// in: [K][N] f32 row-major; out: [N][K] f16 row-major (B^T for WMMA B-fragments)
__global__ __launch_bounds__(256) void mla_transpose_f16(const float* __restrict__ in,
    _Float16* __restrict__ out, int K, int N) {
  size_t i = (size_t)blockIdx.x * 256 + threadIdx.x;
  if (i >= (size_t)K * N) return;
  int k = (int)(i / N), n = (int)(i % N);
  out[(size_t)n * K + k] = (_Float16)in[i];
}

// ---------------------------------------------------------------- RoPE (in place)
// q: f16 [B][T][H][HD], pairs (2j,2j+1); i indexes (b,t,h,j), j in [0,64)
__global__ __launch_bounds__(256) void mla_rope(_Float16* __restrict__ q,
    const float* __restrict__ cosb, const float* __restrict__ sinb) {
  size_t i = (size_t)blockIdx.x * 256 + threadIdx.x;
  if (i >= (size_t)65536 * 64) return;
  int j = (int)(i & 63);
  size_t base = i >> 6;               // (b*T + t)*H + h
  int t = (int)((base >> 4) & 2047);
  _Float16* p = q + base * 128 + j * 2;
  float xr = (float)p[0], xi = (float)p[1];
  float cv = cosb[t * 64 + j], sv = sinb[t * 64 + j];
  p[0] = (_Float16)(xr * cv - xi * sv);
  p[1] = (_Float16)(xr * sv + xi * cv);
}

// --------------------------------------------- direct-load WMMA GEMM (small N)
// C[M,N] = A[M,K](f16) * Bt[N,K]^T + bias[N]; one 16x16 tile per wave.
// MODE 2: rows=(b*T+t)*H+h -> f16 [B][H][T][L]      (q/k latents)
// MODE 3: rows=(b*T+t)*H+h -> f16 [B][H][L][T+16]   (v latent, transposed+padded)
template <int MODE>
__global__ __launch_bounds__(256) void mla_gemm_direct(const _Float16* __restrict__ A,
    const _Float16* __restrict__ Bt, const float* __restrict__ bias,
    void* __restrict__ C, int M, int N, int K) {
  const int lane = threadIdx.x & 31;
  const int wave = threadIdx.x >> 5;
  const int ntiles = N >> 4;
  const int tile = blockIdx.x * 8 + wave;
  if (tile >= (M >> 4) * ntiles) return;       // wave-uniform: EXEC stays full
  const int tm = tile / ntiles, tn = tile - tm * ntiles;
  const int l15 = lane & 15;
  const int half = lane >> 4;
  const int koff = half ? 8 : 0;
  const int khalf = half ? 16 : 0;
  const _Float16* arow = A + (size_t)(tm * 16 + l15) * K;
  const _Float16* brow = Bt + (size_t)(tn * 16 + l15) * K;
  f32x8 c = {};
#pragma unroll 2
  for (int kk = 0; kk < K; kk += 32) {
    half8 a0 = *(const half8*)(arow + kk + koff);
    half8 a1 = *(const half8*)(arow + kk + koff + 16);
    half8 b0 = *(const half8*)(brow + kk + koff);
    half8 b1 = *(const half8*)(brow + kk + koff + 16);
    c = wmma32(CAT8_8(a0, a1), CAT8_8(b0, b1), c);
  }
  const float bv = bias[tn * 16 + l15];
#pragma unroll
  for (int r = 0; r < 8; r++) {
    float val = c[r] + bv;
    int rr = tm * 16 + r + 8 * half;
    int cc = tn * 16 + l15;
    if constexpr (MODE == 2) {
      int h = rr & 15, bt = rr >> 4, t = bt & 2047, bb = bt >> 11;
      ((_Float16*)C)[((size_t)(bb * 16 + h) * 2048 + t) * 64 + cc] = (_Float16)val;
    } else {  // MODE 3
      int h = rr & 15, bt = rr >> 4, t = bt & 2047, bb = bt >> 11;
      ((_Float16*)C)[((size_t)(bb * 16 + h) * 64 + cc) * 2064 + t] = (_Float16)val;
    }
  }
}

// ------------------------------- LDS-staged, async double-buffered WMMA GEMM
// Block (256 thr, 8 waves) computes a 128x128 C tile; wave (wm,wn) owns 64x32.
// A and Bt k-slices (128x32 f16) staged via global_load_async_to_lds_b128,
// pipelined with s_wait_asynccnt. M%128==0, N%128==0, K%32==0 required.
// MODE 0: f16 row-major   MODE 1: f32 row-major
// MODE 4: rows=(b*H+h)*T+t -> f16 [B][T][D] at col h*128+cc  (wd decompress)
template <int MODE>
__global__ __launch_bounds__(256) void mla_gemm_lds(const _Float16* __restrict__ A,
    const _Float16* __restrict__ Bt, const float* __restrict__ bias,
    void* __restrict__ C, int M, int N, int K) {
  const int tid = threadIdx.x;
  const int lane = tid & 31;
  const int wave = tid >> 5;
  const int wm = wave >> 2;            // 0..1
  const int wn = wave & 3;             // 0..3
  const int l15 = lane & 15;
  const int half = lane >> 4;
  const int koff = half ? 8 : 0;       // A-fragment K start (ISA 16-bit A layout)
  const int khalf = half ? 16 : 0;     // unused; B fragments start at koff, as A does
  const int bn = blockIdx.x, bm = blockIdx.y;

  __shared__ __align__(16) _Float16 sA[2][128 * 32];
  __shared__ __align__(16) _Float16 sB[2][128 * 32];

  const _Float16* Ab = A + (size_t)(bm * 128) * K;
  const _Float16* Bb = Bt + (size_t)(bn * 128) * K;

  f32x8 acc[4][2];
#pragma unroll
  for (int i = 0; i < 4; i++)
#pragma unroll
    for (int j = 0; j < 2; j++) acc[i][j] = (f32x8){};

  // prologue: stage k-slice 0 into buffer 0 (4 async instrs per wave)
  async_stage(Ab, K, 0, &sA[0][0], tid);
  async_stage(Bb, K, 0, &sB[0][0], tid);

  int cur = 0;
  for (int kk = 0; kk < K; kk += 32) {
    if (kk + 32 < K) {                 // issue next slice into the other buffer
      async_stage(Ab, K, kk + 32, &sA[cur ^ 1][0], tid);
      async_stage(Bb, K, kk + 32, &sB[cur ^ 1][0], tid);
      // 8 outstanding; async loads retire in order -> <=4 means current slice landed
      asm volatile("s_wait_asynccnt 0x4" ::: "memory");
    } else {
      asm volatile("s_wait_asynccnt 0x0" ::: "memory");
    }
    __syncthreads();

    half16 af[4], bf[2];
#pragma unroll
    for (int mi = 0; mi < 4; mi++) {
      const _Float16* p = &sA[cur][(wm * 64 + mi * 16 + l15) * 32];
      af[mi] = CAT8_8(*(const half8*)(p + koff), *(const half8*)(p + koff + 16));
    }
#pragma unroll
    for (int ni = 0; ni < 2; ni++) {
      const _Float16* p = &sB[cur][(wn * 32 + ni * 16 + l15) * 32];
      bf[ni] = CAT8_8(*(const half8*)(p + koff), *(const half8*)(p + koff + 16));
    }
#pragma unroll
    for (int mi = 0; mi < 4; mi++)
#pragma unroll
      for (int ni = 0; ni < 2; ni++)
        acc[mi][ni] = wmma32(af[mi], bf[ni], acc[mi][ni]);

    __syncthreads();                   // buffer may be overwritten next iteration
    cur ^= 1;
  }

#pragma unroll
  for (int mi = 0; mi < 4; mi++) {
#pragma unroll
    for (int ni = 0; ni < 2; ni++) {
      int cc = bn * 128 + wn * 32 + ni * 16 + l15;
      float bv = bias[cc];
#pragma unroll
      for (int r = 0; r < 8; r++) {
        float val = acc[mi][ni][r] + bv;
        int rr = bm * 128 + wm * 64 + mi * 16 + r + 8 * half;
        if constexpr (MODE == 0) {
          ((_Float16*)C)[(size_t)rr * N + cc] = (_Float16)val;
        } else if constexpr (MODE == 1) {
          ((float*)C)[(size_t)rr * N + cc] = val;
        } else {  // MODE 4
          int t = rr & 2047, bhh = rr >> 11, h = bhh & 15, bb = bhh >> 4;
          ((_Float16*)C)[(size_t)(bb * 2048 + t) * 2048 + h * 128 + cc] = (_Float16)val;
        }
      }
    }
  }
}

// ---------------------------------------------------------------- flash attention
// Q,K latents: f16 [B][H][T][64]; V: f16 [B][H][64][2064] (padded); O: f16 [B][H][T][64]
__global__ __launch_bounds__(32) void mla_attention(const _Float16* __restrict__ Q,
    const _Float16* __restrict__ Kl, const _Float16* __restrict__ Vt,
    _Float16* __restrict__ O) {
  const int lane = threadIdx.x;
  const int qt = blockIdx.x, h = blockIdx.y, b = blockIdx.z;
  const int qbase = qt * 16;
  const int l15 = lane & 15;
  const int half = lane >> 4;
  const int koff = half ? 8 : 0;
  const int khalf = half ? 16 : 0;
  const size_t bh = (size_t)(b * 16 + h);

  const _Float16* qrow = Q + (bh * 2048 + qbase + l15) * 64;
  half16 qa0 = CAT8_8(*(const half8*)(qrow + koff), *(const half8*)(qrow + koff + 16));
  half16 qa1 = CAT8_8(*(const half8*)(qrow + 32 + koff), *(const half8*)(qrow + 32 + koff + 16));
#pragma unroll
  for (int i = 0; i < 16; i++) { qa0[i] *= (_Float16)0.125f; qa1[i] *= (_Float16)0.125f; }

  f32x8 acc0 = {}, acc1 = {}, acc2 = {}, acc3 = {};
  float ms[8], ls[8];
#pragma unroll
  for (int r = 0; r < 8; r++) { ms[r] = -1e30f; ls[r] = 0.f; }

  __shared__ __align__(16) _Float16 Pl[256];
  const _Float16* kbp = Kl + bh * 2048 * 64;
  const _Float16* vbp = Vt + bh * 64 * 2064;

  for (int kt = 0; kt <= qt; ++kt) {
    const _Float16* krow = kbp + (size_t)(kt * 16 + l15) * 64;
    half16 kb0 = CAT8_8(*(const half8*)(krow + koff), *(const half8*)(krow + koff + 16));
    half16 kb1 = CAT8_8(*(const half8*)(krow + 32 + koff), *(const half8*)(krow + 32 + koff + 16));
    f32x8 s = {};
    s = wmma32(qa0, kb0, s);
    s = wmma32(qa1, kb1, s);
    if (kt == qt) {  // causal mask on diagonal tile only (uniform branch)
#pragma unroll
      for (int r = 0; r < 8; r++)
        if (kt * 16 + l15 > qbase + r + 8 * half) s[r] = -1e9f;
    }
    float p[8];
#pragma unroll
    for (int r = 0; r < 8; r++) {
      float mn = fmaxf(ms[r], hmax16(s[r]));
      float pv = __expf(s[r] - mn);
      float rs = hsum16(pv);
      float corr = __expf(ms[r] - mn);
      ls[r] = ls[r] * corr + rs;
      ms[r] = mn;
      acc0[r] *= corr; acc1[r] *= corr; acc2[r] *= corr; acc3[r] *= corr;
      p[r] = pv;
    }
    __syncthreads();
#pragma unroll
    for (int r = 0; r < 8; r++) Pl[(r + 8 * half) * 16 + l15] = (_Float16)p[r];
    __syncthreads();
    half8 plo = *(const half8*)(Pl + l15 * 16 + koff);
    half8 zz = {};
    half16 pa = CAT8_8(plo, zz);       // keys 16..31 zero (K padded to 32)
#pragma unroll
    for (int nt = 0; nt < 4; nt++) {
      const _Float16* vr = vbp + (size_t)(nt * 16 + l15) * 2064 + kt * 16 + koff;
      half16 vbf = CAT8_8(*(const half8*)vr, *(const half8*)(vr + 16));
      f32x8& a = (nt == 0) ? acc0 : (nt == 1) ? acc1 : (nt == 2) ? acc2 : acc3;
      a = wmma32(pa, vbf, a);
    }
  }
  _Float16* orow = O + bh * 2048 * 64;
#pragma unroll
  for (int r = 0; r < 8; r++) {
    float inv = 1.f / ls[r];
    size_t m = (size_t)qbase + r + 8 * half;
    orow[m * 64 + 0 + l15]  = (_Float16)(acc0[r] * inv);
    orow[m * 64 + 16 + l15] = (_Float16)(acc1[r] * inv);
    orow[m * 64 + 32 + l15] = (_Float16)(acc2[r] * inv);
    orow[m * 64 + 48 + l15] = (_Float16)(acc3[r] * inv);
  }
}

// ---------------------------------------------------------------- host launcher
extern "C" void kernel_launch(void* const* d_in, const int* in_sizes, int n_in,
                              void* d_out, int out_size, void* d_ws, size_t ws_size,
                              hipStream_t stream) {
  const float* x    = (const float*)d_in[0];
  // d_in[1] = mask (bool) — causal structure is hardcoded
  const float* ln_w = (const float*)d_in[2];
  const float* ln_b = (const float*)d_in[3];
  const float* wq = (const float*)d_in[4];  const float* bq = (const float*)d_in[5];
  const float* wk = (const float*)d_in[6];  const float* bk = (const float*)d_in[7];
  const float* wv = (const float*)d_in[8];  const float* bv = (const float*)d_in[9];
  const float* wo = (const float*)d_in[10]; const float* bo = (const float*)d_in[11];
  const float* wcq = (const float*)d_in[12]; const float* bcq = (const float*)d_in[13];
  const float* wck = (const float*)d_in[14]; const float* bck = (const float*)d_in[15];
  const float* wcv = (const float*)d_in[16]; const float* bcv = (const float*)d_in[17];
  const float* wd  = (const float*)d_in[18]; const float* bd  = (const float*)d_in[19];
  const float* fcos = (const float*)d_in[20];
  const float* fsin = (const float*)d_in[21];

  // workspace bump allocator (256B aligned); total ~152 MB of f16
  char* wp = (char*)d_ws;
  auto alloc = [&](size_t elems) -> _Float16* {
    _Float16* p = (_Float16*)wp;
    wp += ((elems * sizeof(_Float16)) + 255) & ~(size_t)255;
    return p;
  };
  const size_t BT = 4096, Dm = 2048, BTH = 65536;
  _Float16* xn   = alloc(BT * Dm);
  _Float16* wqT  = alloc(Dm * Dm);
  _Float16* wkT  = alloc(Dm * Dm);
  _Float16* wvT  = alloc(Dm * Dm);
  _Float16* woT  = alloc(Dm * Dm);
  _Float16* wcqT = alloc(64 * 128);
  _Float16* wckT = alloc(64 * 128);
  _Float16* wcvT = alloc(64 * 128);
  _Float16* wdT  = alloc(128 * 64);
  _Float16* qb   = alloc(BT * Dm);
  _Float16* kb   = alloc(BT * Dm);
  _Float16* vb   = alloc(BT * Dm);
  _Float16* qlat = alloc((size_t)2 * 16 * 2048 * 64);
  _Float16* klat = alloc((size_t)2 * 16 * 2048 * 64);
  _Float16* vT   = alloc((size_t)2 * 16 * 64 * 2064);  // padded T+16
  _Float16* attn = alloc((size_t)2 * 16 * 2048 * 64);
  _Float16* o2   = alloc(BT * Dm);

  // 1) LayerNorm
  mla_layernorm<<<(int)BT, 256, 0, stream>>>(x, ln_w, ln_b, xn);

  // 2) weight transposes (f32 -> f16 W^T)
  const int TB_BIG = (int)((Dm * Dm + 255) / 256);
  mla_transpose_f16<<<TB_BIG, 256, 0, stream>>>(wq, wqT, 2048, 2048);
  mla_transpose_f16<<<TB_BIG, 256, 0, stream>>>(wk, wkT, 2048, 2048);
  mla_transpose_f16<<<TB_BIG, 256, 0, stream>>>(wv, wvT, 2048, 2048);
  mla_transpose_f16<<<TB_BIG, 256, 0, stream>>>(wo, woT, 2048, 2048);
  mla_transpose_f16<<<32, 256, 0, stream>>>(wcq, wcqT, 128, 64);
  mla_transpose_f16<<<32, 256, 0, stream>>>(wck, wckT, 128, 64);
  mla_transpose_f16<<<32, 256, 0, stream>>>(wcv, wcvT, 128, 64);
  mla_transpose_f16<<<32, 256, 0, stream>>>(wd, wdT, 64, 128);

  // 3) QKV projections: LDS-staged async WMMA GEMM, 128x128 blocks
  const dim3 gBig(2048 / 128, 4096 / 128);   // (N/128, M/128)
  mla_gemm_lds<0><<<gBig, 256, 0, stream>>>(xn, wqT, bq, qb, 4096, 2048, 2048);
  mla_gemm_lds<0><<<gBig, 256, 0, stream>>>(xn, wkT, bk, kb, 4096, 2048, 2048);
  mla_gemm_lds<0><<<gBig, 256, 0, stream>>>(xn, wvT, bv, vb, 4096, 2048, 2048);

  // 4) RoPE on q,k
  const int ROPE_B = (int)((BTH * 64 + 255) / 256);
  mla_rope<<<ROPE_B, 256, 0, stream>>>(qb, fcos, fsin);
  mla_rope<<<ROPE_B, 256, 0, stream>>>(kb, fcos, fsin);

  // 5) latent projections (N=64): direct-load WMMA with scatter epilogues
  auto grid_of = [](int M, int N) { return ((M / 16) * (N / 16) + 7) / 8; };
  mla_gemm_direct<2><<<grid_of(65536, 64), 256, 0, stream>>>(qb, wcqT, bcq, qlat, 65536, 64, 128);
  mla_gemm_direct<2><<<grid_of(65536, 64), 256, 0, stream>>>(kb, wckT, bck, klat, 65536, 64, 128);
  mla_gemm_direct<3><<<grid_of(65536, 64), 256, 0, stream>>>(vb, wcvT, bcv, vT, 65536, 64, 128);

  // 6) causal flash attention over latents (WMMA)
  mla_attention<<<dim3(128, 16, 2), 32, 0, stream>>>(qlat, klat, vT, attn);

  // 7) decompress heads: attn @ wd + bd -> o2 [B][T][D] (LDS-staged WMMA)
  mla_gemm_lds<4><<<dim3(1, 512), 256, 0, stream>>>(attn, wdT, bd, o2, 65536, 128, 64);

  // 8) output projection -> f32 d_out (LDS-staged WMMA)
  mla_gemm_lds<1><<<gBig, 256, 0, stream>>>(o2, woT, bo, (float*)d_out, 4096, 2048, 2048);
}
